// FCOSOutputs_70334384439404
// MI455X (gfx1250) — compile-verified, matrix operand order checked by reading, not run on hardware
//
#include <hip/hip_runtime.h>
#include <hip/hip_bf16.h>
#include <math.h>

typedef __attribute__((ext_vector_type(16))) _Float16 v16h;
typedef __attribute__((ext_vector_type(8)))  float    v8f;

#define NUM_CLASSES 80
#define INF_VAL 100000000.0f
#define RADIUS 1.5f
#define TPB 256
#define MAXG 64

#if __has_builtin(__builtin_amdgcn_global_load_async_to_lds_b128) && \
    __has_builtin(__builtin_amdgcn_global_load_async_to_lds_b32)  && \
    __has_builtin(__builtin_amdgcn_s_wait_asynccnt)
#define USE_ASYNC_LDS 1
// Pointee types matching the builtin prototypes (from hipcc diagnostics):
//   b128: int __attribute__((vector_size(16))) in AS1 / AS3
//   b32 : int in AS1 / AS3
typedef int i32x4_t __attribute__((vector_size(16)));
typedef __attribute__((address_space(1))) i32x4_t* gas_b128_t;
typedef __attribute__((address_space(3))) i32x4_t* lds_b128_t;
typedef __attribute__((address_space(1))) int*     gas_b32_t;
typedef __attribute__((address_space(3))) int*     lds_b32_t;
// Integer round-trips sidestep cross-AS pointer-cast restrictions:
// global AS1 values == flat values; AS3 offset == low 32 bits of the flat
// LDS-aperture address (ISA 10.2).
#define TO_GLOBAL_B128(p) ((gas_b128_t)(size_t)(p))
#define TO_GLOBAL_B32(p)  ((gas_b32_t)(size_t)(p))
#define TO_LDS_B128(p)    ((lds_b128_t)(unsigned int)(size_t)(p))
#define TO_LDS_B32(p)     ((lds_b32_t)(unsigned int)(size_t)(p))
#else
#define USE_ASYNC_LDS 0
#endif

// ---------------------------------------------------------------------------
// Wave-level sum reduction through the WMMA pipe (wave32).
// x = hi + lo in f16 (compensated split), both placed in the A matrix
// (rowsum semantics make the slot layout irrelevant), B = ones.
// D[m,n] = rowsum_m; sum over M of column 0 (lanes 0 and 16, VGPRs 0..7 of D)
// equals the f32-accurate lane sum. EXEC must be all ones at the call site.
// ---------------------------------------------------------------------------
__device__ __forceinline__ float wave_reduce_wmma(float x) {
    v16h a = {};
    _Float16 hi = (_Float16)x;
    _Float16 lo = (_Float16)(x - (float)hi);
    a[0] = hi;
    a[1] = lo;
    v16h b;
#pragma unroll
    for (int i = 0; i < 16; ++i) b[i] = (_Float16)1.0f;
    v8f c = {};
    c = __builtin_amdgcn_wmma_f32_16x16x32_f16(
        /*neg_a=*/false, a, /*neg_b=*/false, b,
        /*c_mod=*/(short)0, c, /*reuse_a=*/false, /*reuse_b=*/false);
    float s = c[0] + c[1] + c[2] + c[3] + c[4] + c[5] + c[6] + c[7];
    // D column 0: lanes 0 (M=0..7) and 16 (M=8..15).
    return __shfl(s, 0, 32) + __shfl(s, 16, 32);
}

__device__ __forceinline__ float focal_term(float x, bool tgt) {
    float e  = __expf(-fabsf(x));
    float ce = fmaxf(x, 0.0f) - (tgt ? x : 0.0f) + log1pf(e);
    float p  = (x >= 0.0f) ? (1.0f / (1.0f + e)) : (e / (1.0f + e));
    float pt = tgt ? p : (1.0f - p);
    float at = tgt ? 0.25f : 0.75f;
    float om = 1.0f - pt;
    return ce * om * om * at;
}

// ---------------------------------------------------------------------------
// Main kernel: one block = 256 locations of one batch image.
// Phase 0: async-stage gt boxes/classes (global -> LDS, ASYNCcnt-tracked).
// Phase 1: box assignment, GIoU + ctrness losses.
// Phase 2: focal loss over the (256 x 80) logits tile, float4-coalesced.
// Reduction: per-wave WMMA reduce -> per-block partials to d_ws (no atomics).
// ---------------------------------------------------------------------------
__global__ void __launch_bounds__(TPB)
fcos_main_kernel(const float* __restrict__ logits,
                 const float* __restrict__ reg_pred,
                 const float* __restrict__ ctrness,
                 const float* __restrict__ locations,
                 const int*   __restrict__ level_ids,
                 const float* __restrict__ gt_boxes,
                 const int*   __restrict__ gt_classes,
                 float* __restrict__ block_partials,
                 int L, int G) {
    __shared__ alignas(16) float sraw[MAXG * 4];
    __shared__ int   scls[MAXG];
    __shared__ float sbx0[MAXG], sby0[MAXG], sbx1[MAXG], sby1[MAXG];
    __shared__ float scx[MAXG], scy[MAXG], sarea[MAXG];
    __shared__ int   slab[TPB];
    __shared__ float wsum[8][5];

    const int tid = threadIdx.x;
    const int b   = blockIdx.y;
    const int l0  = blockIdx.x * TPB;
    const int l   = l0 + tid;

    // ---- Phase 0: stage raw gt data into LDS -----------------------------
#if USE_ASYNC_LDS
    if (tid < G) {
        // One 16B row per lane; rows are 16B-aligned (G*4*4 = 800 = 50*16).
        __builtin_amdgcn_global_load_async_to_lds_b128(
            TO_GLOBAL_B128(gt_boxes + ((size_t)b * G + tid) * 4),
            TO_LDS_B128(&sraw[tid * 4]), 0, 0);
        __builtin_amdgcn_global_load_async_to_lds_b32(
            TO_GLOBAL_B32(gt_classes + (size_t)b * G + tid),
            TO_LDS_B32(&scls[tid]), 0, 0);
    }
    __builtin_amdgcn_s_wait_asynccnt(0);
#else
    if (tid < G) {
        float4 v = ((const float4*)gt_boxes)[(size_t)b * G + tid];
        sraw[tid * 4 + 0] = v.x; sraw[tid * 4 + 1] = v.y;
        sraw[tid * 4 + 2] = v.z; sraw[tid * 4 + 3] = v.w;
        scls[tid] = gt_classes[(size_t)b * G + tid];
    }
#endif
    __syncthreads();
    if (tid < G) {
        float x0 = sraw[tid * 4 + 0], y0 = sraw[tid * 4 + 1];
        float x1 = sraw[tid * 4 + 2], y1 = sraw[tid * 4 + 3];
        sbx0[tid] = x0; sby0[tid] = y0; sbx1[tid] = x1; sby1[tid] = y1;
        scx[tid]  = (x0 + x1) * 0.5f;
        scy[tid]  = (y0 + y1) * 0.5f;
        sarea[tid] = (x1 - x0) * (y1 - y0);
    }
    __syncthreads();

    // ---- Phase 1: assignment + GIoU + ctrness ----------------------------
    float part_focal = 0.0f, part_reg = 0.0f, part_ctrw = 0.0f;
    float part_bce = 0.0f, part_pos = 0.0f;
    int   label = NUM_CLASSES;

    if (l < L) {
        float xs = locations[(size_t)l * 2 + 0];
        float ys = locations[(size_t)l * 2 + 1];
        int   lvl = level_ids[l];
        float stride = (float)(8 << lvl);
        float soi_lo = (lvl == 0) ? -1.0f : (float)(32 << lvl);
        float soi_hi = (lvl == 4) ? INF_VAL : (float)(64 << lvl);
        float sr = stride * RADIUS;

        float minArea = INF_VAL;
        float bl = 0.f, bt = 0.f, br = 0.f, bb = 0.f;
        int   bcls = 0;
        for (int g = 0; g < G; ++g) {
            float x0 = sbx0[g], y0 = sby0[g], x1 = sbx1[g], y1 = sby1[g];
            float l_ = xs - x0, t_ = ys - y0, r_ = x1 - xs, b_ = y1 - ys;
            float mreg = fmaxf(fmaxf(l_, t_), fmaxf(r_, b_));
            float cx = scx[g], cy = scy[g];
            float xmin = fmaxf(cx - sr, x0), ymin = fmaxf(cy - sr, y0);
            float xmax = fminf(cx + sr, x1), ymax = fminf(cy + sr, y1);
            float cbm = fminf(fminf(xs - xmin, ys - ymin),
                              fminf(xmax - xs, ymax - ys));
            bool ok = (cbm > 0.0f) && (mreg >= soi_lo) && (mreg <= soi_hi);
            float a = ok ? sarea[g] : INF_VAL;
            if (a < minArea) {
                minArea = a; bl = l_; bt = t_; br = r_; bb = b_; bcls = scls[g];
            }
        }
        bool pos = minArea < INF_VAL;
        label = pos ? bcls : NUM_CLASSES;

        float tl = 1.f, tt = 1.f, tr = 1.f, tb = 1.f;
        if (pos) {
            float inv = 1.0f / stride;
            tl = bl * inv; tt = bt * inv; tr = br * inv; tb = bb * inv;
        }
        float ctr_t = sqrtf((fminf(tl, tr) / fmaxf(tl, tr)) *
                            (fminf(tt, tb) / fmaxf(tt, tb)));
        float ctr_w = pos ? ctr_t : 0.0f;

        float4 rp = ((const float4*)reg_pred)[(size_t)b * L + l];
        float pl = rp.x, pt = rp.y, pr = rp.z, pb = rp.w;
        float t_area = (tl + tr) * (tt + tb);
        float p_area = (pl + pr) * (pt + pb);
        float w_i = fminf(pl, tl) + fminf(pr, tr);
        float h_i = fminf(pt, tt) + fminf(pb, tb);
        float g_w = fmaxf(pl, tl) + fmaxf(pr, tr);
        float g_h = fmaxf(pt, tt) + fmaxf(pb, tb);
        float a_i = w_i * h_i;
        float a_c = g_w * g_h;
        float a_u = t_area + p_area - a_i;
        float ious  = (a_i + 1.0f) / (a_u + 1.0f);
        float gious = ious - (a_c - a_u) / fmaxf(a_c, 1e-7f);
        part_reg  = (1.0f - gious) * ctr_w;
        part_ctrw = ctr_w;

        float ctr = ctrness[(size_t)b * L + l];
        if (pos) {
            float e = __expf(-fabsf(ctr));
            part_bce = fmaxf(ctr, 0.0f) - ctr * ctr_w + log1pf(e);
            part_pos = 1.0f;
        }
    }
    slab[tid] = label;
    __syncthreads();

    // ---- Phase 2: focal loss over the (TPB x 80) logits tile -------------
    const float4* lg =
        (const float4*)logits + ((size_t)b * L + (size_t)l0) * 20;
#pragma unroll 4
    for (int k = 0; k < 20; ++k) {
        int f4  = tid + k * TPB;
        int li  = f4 / 20;                // location in tile
        int c0  = (f4 - li * 20) * 4;     // first class of this float4
        __builtin_prefetch(((const float*)(lg + f4)) + 4 * TPB, 0, 3);
        if (l0 + li < L) {
            float4 v = lg[f4];
            int lab = slab[li];
            part_focal += focal_term(v.x, lab == c0 + 0);
            part_focal += focal_term(v.y, lab == c0 + 1);
            part_focal += focal_term(v.z, lab == c0 + 2);
            part_focal += focal_term(v.w, lab == c0 + 3);
        }
    }

    // ---- Block reduction via WMMA (all threads converged; EXEC all-1s) ---
    float r0 = wave_reduce_wmma(part_focal);
    float r1 = wave_reduce_wmma(part_reg);
    float r2 = wave_reduce_wmma(part_ctrw);
    float r3 = wave_reduce_wmma(part_bce);
    float r4 = wave_reduce_wmma(part_pos);

    int wave = tid >> 5;
    if ((tid & 31) == 0) {
        wsum[wave][0] = r0; wsum[wave][1] = r1; wsum[wave][2] = r2;
        wsum[wave][3] = r3; wsum[wave][4] = r4;
    }
    __syncthreads();
    if (tid == 0) {
        float acc[5] = {0.f, 0.f, 0.f, 0.f, 0.f};
        for (int w = 0; w < 8; ++w)
            for (int j = 0; j < 5; ++j) acc[j] += wsum[w][j];
        float* out = block_partials + (size_t)(blockIdx.y * gridDim.x + blockIdx.x) * 8;
        for (int j = 0; j < 5; ++j) out[j] = acc[j];
    }
}

// ---------------------------------------------------------------------------
// Finalize: single block sums per-block partials in a fixed order
// (deterministic) and produces the 3 losses.
// ---------------------------------------------------------------------------
__global__ void __launch_bounds__(TPB)
fcos_finalize_kernel(const float* __restrict__ block_partials, int nblocks,
                     float* __restrict__ out) {
    __shared__ float sm[5][TPB];
    float s[5] = {0.f, 0.f, 0.f, 0.f, 0.f};
    for (int i = threadIdx.x; i < nblocks; i += TPB) {
        const float* p = block_partials + (size_t)i * 8;
        for (int j = 0; j < 5; ++j) s[j] += p[j];
    }
    for (int j = 0; j < 5; ++j) sm[j][threadIdx.x] = s[j];
    __syncthreads();
    for (int off = TPB / 2; off > 0; off >>= 1) {
        if (threadIdx.x < off)
            for (int j = 0; j < 5; ++j)
                sm[j][threadIdx.x] += sm[j][threadIdx.x + off];
        __syncthreads();
    }
    if (threadIdx.x == 0) {
        float num_pos = fmaxf(sm[4][0], 1.0f);
        out[0] = sm[0][0] / num_pos;                    // loss_cls
        out[1] = sm[1][0] / fmaxf(sm[2][0], 1e-6f);     // loss_reg
        out[2] = sm[3][0] / num_pos;                    // loss_ctr
    }
}

extern "C" void kernel_launch(void* const* d_in, const int* in_sizes, int n_in,
                              void* d_out, int out_size, void* d_ws, size_t ws_size,
                              hipStream_t stream) {
    const float* logits    = (const float*)d_in[0];
    const float* reg_pred  = (const float*)d_in[1];
    const float* ctrness   = (const float*)d_in[2];
    const float* locations = (const float*)d_in[3];
    const int*   level_ids = (const int*)d_in[4];
    const float* gt_boxes  = (const float*)d_in[5];
    const int*   gt_classes= (const int*)d_in[6];

    const int L = in_sizes[3] / 2;          // locations is (L, 2)
    const int B = in_sizes[2] / L;          // ctrness is (B, L)
    const int G = in_sizes[6] / B;          // gt_classes is (B, G)

    const int chunks = (L + TPB - 1) / TPB;
    float* bp = (float*)d_ws;               // chunks*B blocks * 8 floats

    dim3 grid(chunks, B);
    fcos_main_kernel<<<grid, TPB, 0, stream>>>(
        logits, reg_pred, ctrness, locations, level_ids,
        gt_boxes, gt_classes, bp, L, G);
    fcos_finalize_kernel<<<1, TPB, 0, stream>>>(bp, chunks * B, (float*)d_out);
}
